// SwinTransformerBlock_13202729468049
// MI455X (gfx1250) — compile-verified, matrix operand order checked by reading, not run on hardware
//
#include <hip/hip_runtime.h>
#include <hip/hip_bf16.h>

// ---------------------------------------------------------------------------
// Swin Transformer block on gfx1250 (MI455X), wave32 + WMMA f16 + TDM.
//
// Pipeline (f16 staging, f32 accumulate via v_wmma_f32_16x16x32_f16):
//   K1 cvt      : weights fp32 -> f16
//   K2 ln_win   : LN1 + roll(-3,-3) + window partition -> xwin[2048][64][384] f16
//   K3 gemm<0>  : xwin @ qkv_w + qkv_b (q scaled)      -> qkv [2048][64][1152] f16
//   K4 attn     : per (window, head): softmax(q k^T + bias + mask) v  -> aout f16
//   K5 gemm<1>  : aout @ proj_w + proj_b, window-reverse + roll + resid -> x2 f32
//   K6 ln       : LN2(x2) -> h2 f16
//   K7 gemm<2>  : h2 @ fc1_w + b, exact GELU -> hid f16
//   K8 gemm<3>  : hid @ fc2_w + b + x2 -> d_out f32
//
// Data movement: GEMM tiles and attention q/k/v tiles are staged into LDS by
// the Tensor Data Mover (tensor_load_to_lds, TENSORcnt) issued by wave 0 of
// each workgroup, double-buffered in the GEMM so DMA overlaps WMMA compute.
// The TDM descriptor's pad_interval/pad_amount add 4 DWORDs of padding per
// LDS row to reproduce the bank-conflict-avoiding padded row strides.
// Fallback (builtin unavailable): synchronous per-thread b128 loads.
//
// WMMA fragment layout assumptions (ISA 7.12.2):
//   A (16x32 f16): lane = hi*16 + r; row = r;
//       element e: K = hi*8 + e (e<8), 16 + hi*8 + e-8 (e>=8)
//   B (32x16 f16): mirror of A with row -> column (col = r, same K map).
//   C/D (16x16 f32): element i: row = hi*8 + i, col = r (documented).
// ---------------------------------------------------------------------------

typedef __attribute__((ext_vector_type(16))) _Float16 v16h;
typedef __attribute__((ext_vector_type(8)))  float    v8f;
typedef __attribute__((ext_vector_type(4)))  unsigned int v4u;
typedef __attribute__((ext_vector_type(8)))  int      v8i;
typedef __attribute__((ext_vector_type(4)))  int      v4i;

#define HH 56
#define WW_IMG 56
#define WS 7
#define SHIFT 3
#define DIM 384
#define HEADS 12
#define HD 32
#define NTOK 49
#define NPAD 64
#define NWIN 2048          // B * NW = 32 * 64
#define LSEQ 3136
#define MLP_H 1536
#define QKV_N 1152
#define BL 100352          // BATCH * LSEQ

#if defined(__HIP_DEVICE_COMPILE__) && \
    __has_builtin(__builtin_amdgcn_tensor_load_to_lds) && \
    __has_builtin(__builtin_amdgcn_s_wait_tensorcnt)
#define HAVE_TDM 1
#else
#define HAVE_TDM 0
#endif

__device__ __forceinline__ int kmap(int e, int hi) {
    return (e < 8) ? (hi * 8 + e) : (8 + hi * 8 + e); // e>=8: 16 + hi*8 + (e-8)
}

#if HAVE_TDM
typedef __attribute__((address_space(3))) char lds_char;
__device__ __forceinline__ unsigned ldsOff(const void* p) {
    return (unsigned)(unsigned long long)(lds_char*)p;   // addrspacecast -> LDS byte offset
}

// 2D TDM tile load: tile_rows x tile_cols f16 elements, global row stride
// `stride0` elements, destination LDS rows padded by (padA+1) DWORDs every
// 2^(padI+1) DWORDs (padI/padA < 0 disables padding).
__device__ __forceinline__ void tdm_load_2d(
    const _Float16* gptr, unsigned lds_byte_off,
    int dim0_elems, int dim1_rows,
    int tile_cols, int tile_rows,
    int stride0_elems, int padI, int padA)
{
    unsigned long long ga = (unsigned long long)gptr;
    v4u g0;
    g0[0] = 1u;                                             // count=1 (valid D#)
    g0[1] = lds_byte_off;                                   // lds_addr
    g0[2] = (unsigned)(ga & 0xFFFFFFFFu);                   // global_addr[31:0]
    g0[3] = (unsigned)((ga >> 32) & 0x1FFFFFFu) | (2u << 30); // addr[56:32] | type=2
    unsigned w0 = (1u << 16);                               // data_size=1 (2 bytes)
    if (padI >= 0)
        w0 |= (1u << 20) | ((unsigned)padI << 22) | ((unsigned)padA << 25);
    v8i g1;
    g1[0] = (int)w0;
    g1[1] = (int)(((unsigned)dim0_elems & 0xFFFFu) << 16);                 // dim0[15:0]
    g1[2] = (int)((((unsigned)dim0_elems >> 16) & 0xFFFFu) |
                  (((unsigned)dim1_rows & 0xFFFFu) << 16));                // dim0[31:16]|dim1[15:0]
    g1[3] = (int)((((unsigned)dim1_rows >> 16) & 0xFFFFu) |
                  ((unsigned)tile_cols << 16));                            // dim1[31:16]|tile0
    g1[4] = (int)(unsigned)tile_rows;                                      // tile1 | tile2=0
    g1[5] = (int)(unsigned)stride0_elems;                                  // dim0_stride[31:0]
    g1[6] = 0;
    g1[7] = 0;
    v4i z = {0, 0, 0, 0};
#if __clang_major__ >= 23
    v8i z8 = {0, 0, 0, 0, 0, 0, 0, 0};
    __builtin_amdgcn_tensor_load_to_lds(g0, g1, z, z, z8, 0);
#else
    __builtin_amdgcn_tensor_load_to_lds(g0, g1, z, z, 0);
#endif
}
#endif // HAVE_TDM

// --------------------------- fp32 -> f16 convert ---------------------------
__global__ void cvt_f32_f16(const float* __restrict__ s, _Float16* __restrict__ d, int n) {
    int i = blockIdx.x * 256 + threadIdx.x;
    if (i < n) d[i] = (_Float16)s[i];
}

// ------------------- LayerNorm (+optional shift/window) --------------------
__global__ __launch_bounds__(256) void ln_kernel(
    const float* __restrict__ x, const float* __restrict__ g,
    const float* __restrict__ bt, _Float16* __restrict__ out,
    int mode, int nrows)
{
    const int wv = threadIdx.x >> 5, lane = threadIdx.x & 31;
    const int row = blockIdx.x * 8 + wv;
    if (row >= nrows) return;
    const int64_t dst = (int64_t)row * DIM;
    const float* src;
    if (mode == 0) {
        const int win = row >> 6, n = row & 63;
        if (n >= NTOK) {
            #pragma unroll
            for (int t = 0; t < 12; ++t) out[dst + lane * 12 + t] = (_Float16)0.f;
            return;
        }
        const int b_ = win >> 6, wg = win & 63, wh = wg >> 3, ww = wg & 7;
        const int i = n / WS, j = n % WS;
        const int hs = (wh * WS + i + SHIFT) % HH;
        const int ws = (ww * WS + j + SHIFT) % WW_IMG;
        src = x + ((int64_t)b_ * LSEQ + hs * WW_IMG + ws) * DIM;
    } else {
        src = x + dst;
    }
    float v[12];
    const float4* s4 = (const float4*)(src + lane * 12);
    float4 a0 = s4[0], a1 = s4[1], a2 = s4[2];
    v[0]=a0.x; v[1]=a0.y; v[2]=a0.z; v[3]=a0.w;
    v[4]=a1.x; v[5]=a1.y; v[6]=a1.z; v[7]=a1.w;
    v[8]=a2.x; v[9]=a2.y; v[10]=a2.z; v[11]=a2.w;
    float s = 0.f, s2 = 0.f;
    #pragma unroll
    for (int t = 0; t < 12; ++t) { s += v[t]; s2 += v[t] * v[t]; }
    #pragma unroll
    for (int d = 16; d >= 1; d >>= 1) {
        s  += __shfl_xor(s,  d, 32);
        s2 += __shfl_xor(s2, d, 32);
    }
    const float mean = s * (1.f / DIM);
    const float var  = s2 * (1.f / DIM) - mean * mean;
    const float rstd = rsqrtf(var + 1e-5f);
    #pragma unroll
    for (int t = 0; t < 12; ++t) {
        const int c = lane * 12 + t;
        out[dst + c] = (_Float16)((v[t] - mean) * rstd * g[c] + bt[c]);
    }
}

// ------------------------------ WMMA GEMM ----------------------------------
// C[M,N] = A[M,K] (f16, lda==K) @ Bw[K,N] (f16 row-major, ldb==N) + bias.
// Block tile 128x64, 8 waves, each 32x32 (2x2 WMMA tiles), K-step 32.
// Double-buffered LDS tiles; wave 0 drives the TDM, all waves compute.
// MODE 0: QKV   (scale first 384 cols by HD^-0.5, store f16 @ ldo)
// MODE 1: PROJ  (window reverse + roll(+3) + residual, store f32)
// MODE 2: FC1   (exact GELU, store f16 @ ldo)
// MODE 3: FC2   (+ residual, store f32)
template <int MODE>
__global__ __launch_bounds__(256) void gemm_wmma(
    const _Float16* __restrict__ A, int lda,
    const _Float16* __restrict__ Bw, int ldb,
    const float* __restrict__ bias,
    _Float16* __restrict__ outH, int ldo,
    float* __restrict__ outF,
    const float* __restrict__ resid,
    int K)
{
    const int tid  = threadIdx.x;
    const int wv   = tid >> 5, lane = tid & 31;
    const int r    = lane & 15, hi = lane >> 4;
    const int rowBlk = blockIdx.y * 128;
    const int colBlk = blockIdx.x * 64;
    const int waveM = (wv >> 1) * 32;
    const int waveN = (wv & 1) * 32;

    __shared__ _Float16 As[2][128][40];  // 80B row stride (16B aligned)
    __shared__ _Float16 Bs[2][32][72];   // 144B row stride (16B aligned)

#if HAVE_TDM
    const unsigned asO[2] = { ldsOff(&As[0][0][0]), ldsOff(&As[1][0][0]) };
    const unsigned bsO[2] = { ldsOff(&Bs[0][0][0]), ldsOff(&Bs[1][0][0]) };
#endif

    auto load_tiles = [&](int kb, int buf) {
#if HAVE_TDM
        if (wv == 0) {
            // A tile: 128 rows x 32 f16; pad every 16 DWORDs by 4 -> 40-halfword rows
            tdm_load_2d(A + (int64_t)rowBlk * lda + kb, asO[buf],
                        K - kb, 128, 32, 128, lda, 3, 3);
            // B tile: 32 rows x 64 f16; pad every 32 DWORDs by 4 -> 72-halfword rows
            tdm_load_2d(Bw + (int64_t)kb * ldb + colBlk, bsO[buf],
                        ldb - colBlk, 32, 64, 32, ldb, 4, 3);
        }
#else
        #pragma unroll
        for (int u = 0; u < 2; ++u) {
            const int c = tid + u * 256;
            const int rr = c >> 2, cc = (c & 3) * 8;
            *(uint4*)(&As[buf][rr][cc]) =
                *(const uint4*)(A + (int64_t)(rowBlk + rr) * lda + kb + cc);
        }
        {
            const int rr = tid >> 3, cc = (tid & 7) * 8;
            *(uint4*)(&Bs[buf][rr][cc]) =
                *(const uint4*)(Bw + (int64_t)(kb + rr) * ldb + colBlk + cc);
        }
        if (kb + 32 < K) {
            __builtin_prefetch(A + (int64_t)(rowBlk + (tid >> 1)) * lda + kb + 32, 0, 1);
            __builtin_prefetch(Bw + (int64_t)(kb + 32 + (tid >> 3)) * ldb + colBlk, 0, 1);
        }
#endif
    };

    v8f acc[2][2] = {};
    const int nk = K / 32;
    load_tiles(0, 0);   // prologue

    for (int t = 0; t < nk; ++t) {
        const int buf = t & 1;
#if HAVE_TDM
        if (wv == 0) __builtin_amdgcn_s_wait_tensorcnt(0);
#endif
        __syncthreads();
        if (t + 1 < nk) load_tiles(32 * (t + 1), buf ^ 1);

        v16h a0, a1, b0, b1;
        #pragma unroll
        for (int e = 0; e < 16; ++e) {
            const int kk = kmap(e, hi);
            a0[e] = As[buf][waveM + r][kk];
            a1[e] = As[buf][waveM + 16 + r][kk];
            b0[e] = Bs[buf][kk][waveN + r];
            b1[e] = Bs[buf][kk][waveN + 16 + r];
        }
        acc[0][0] = __builtin_amdgcn_wmma_f32_16x16x32_f16(false, a0, false, b0, (short)0, acc[0][0], false, false);
        acc[0][1] = __builtin_amdgcn_wmma_f32_16x16x32_f16(false, a0, false, b1, (short)0, acc[0][1], false, false);
        acc[1][0] = __builtin_amdgcn_wmma_f32_16x16x32_f16(false, a1, false, b0, (short)0, acc[1][0], false, false);
        acc[1][1] = __builtin_amdgcn_wmma_f32_16x16x32_f16(false, a1, false, b1, (short)0, acc[1][1], false, false);
#if !HAVE_TDM
        __syncthreads();  // sync path: writes to buf^1 already barriered at top
#endif
    }

    #pragma unroll
    for (int mt = 0; mt < 2; ++mt)
    #pragma unroll
    for (int nt = 0; nt < 2; ++nt)
    #pragma unroll
    for (int i = 0; i < 8; ++i) {
        const int grow = rowBlk + waveM + 16 * mt + hi * 8 + i;
        const int gcol = colBlk + waveN + 16 * nt + r;
        float v = acc[mt][nt][i] + bias[gcol];
        if constexpr (MODE == 0) {
            if (gcol < DIM) v *= 0.17677669529663687f;   // HD^-0.5
            outH[(int64_t)grow * ldo + gcol] = (_Float16)v;
        } else if constexpr (MODE == 1) {
            const int win = grow >> 6, n = grow & 63;
            if (n < NTOK) {
                const int b_ = win >> 6, wg = win & 63, wh = wg >> 3, ww = wg & 7;
                const int ii = n / WS, jj = n % WS;
                const int h = (wh * WS + ii + SHIFT) % HH;
                const int w = (ww * WS + jj + SHIFT) % WW_IMG;
                const int64_t idx = ((int64_t)b_ * LSEQ + h * WW_IMG + w) * DIM + gcol;
                outF[idx] = resid[idx] + v;
            }
        } else if constexpr (MODE == 2) {
            v = 0.5f * v * (1.0f + erff(v * 0.70710678118654752f));
            outH[(int64_t)grow * ldo + gcol] = (_Float16)v;
        } else {
            const int64_t idx = (int64_t)grow * DIM + gcol;
            outF[idx] = resid[idx] + v;
        }
    }
}

// -------------------------- windowed attention -----------------------------
// One block (4 waves) per (window, head). q/k/v tiles staged by TDM.
__global__ __launch_bounds__(128) void attn_kernel(
    const _Float16* __restrict__ qkv, const float* __restrict__ rpb,
    _Float16* __restrict__ aout)
{
    const int bid  = blockIdx.x;
    const int win  = bid / HEADS, head = bid % HEADS;
    const int tid  = threadIdx.x;
    const int wv   = tid >> 5, lane = tid & 31;
    const int r    = lane & 15, hi = lane >> 4;
    const int wg   = win & 63, wh = wg >> 3, ww = wg & 7;

    __shared__ _Float16 Qs[64][40];
    __shared__ _Float16 Ks[64][40];
    __shared__ _Float16 Vs[64][40];
    __shared__ _Float16 Ps[4][16][72];

    const int64_t base = (int64_t)win * NPAD * QKV_N + head * HD;
#if HAVE_TDM
    if (wv == 0) {
        tdm_load_2d(qkv + base,           ldsOff(&Qs[0][0]), QKV_N - head * HD,           64, 32, 64, QKV_N, 3, 3);
        tdm_load_2d(qkv + base + DIM,     ldsOff(&Ks[0][0]), QKV_N - DIM - head * HD,     64, 32, 64, QKV_N, 3, 3);
        tdm_load_2d(qkv + base + 2 * DIM, ldsOff(&Vs[0][0]), QKV_N - 2 * DIM - head * HD, 64, 32, 64, QKV_N, 3, 3);
        __builtin_amdgcn_s_wait_tensorcnt(0);
    }
#else
    #pragma unroll
    for (int u = 0; u < 2; ++u) {
        const int c = tid + u * 128;        // 256 uint4 per matrix
        const int rr = c >> 2, cc = (c & 3) * 8;
        const int64_t rb = base + (int64_t)rr * QKV_N + cc;
        *(uint4*)(&Qs[rr][cc]) = *(const uint4*)(qkv + rb);
        *(uint4*)(&Ks[rr][cc]) = *(const uint4*)(qkv + rb + DIM);
        *(uint4*)(&Vs[rr][cc]) = *(const uint4*)(qkv + rb + 2 * DIM);
    }
#endif
    __syncthreads();

    const int rm = 16 * wv;
    v16h a;
    #pragma unroll
    for (int e = 0; e < 16; ++e) a[e] = Qs[rm + r][kmap(e, hi)];

    float sv[4][8];
    #pragma unroll
    for (int ct = 0; ct < 4; ++ct) {
        v16h bf;
        #pragma unroll
        for (int e = 0; e < 16; ++e) bf[e] = Ks[16 * ct + r][kmap(e, hi)];
        v8f z = {};
        v8f s = __builtin_amdgcn_wmma_f32_16x16x32_f16(false, a, false, bf, (short)0, z, false, false);
        #pragma unroll
        for (int i = 0; i < 8; ++i) sv[ct][i] = s[i];
    }

    #pragma unroll
    for (int i = 0; i < 8; ++i) {
        const int n = rm + hi * 8 + i;
        const int ncl = n < NTOK ? n : NTOK - 1;
        const int i1 = ncl / WS, j1 = ncl % WS;
        const int ph = wh * WS + i1, pw = ww * WS + j1;
        const int reg1 = (ph < 49 ? 0 : (ph < 53 ? 1 : 2)) * 3
                       + (pw < 49 ? 0 : (pw < 53 ? 1 : 2));
        float vals[4];
        #pragma unroll
        for (int ct = 0; ct < 4; ++ct) {
            const int m = 16 * ct + r;
            if (m < NTOK) {
                const int i2 = m / WS, j2 = m % WS;
                const int ridx = (i1 - i2 + WS - 1) * (2 * WS - 1) + (j1 - j2 + WS - 1);
                const float bv = rpb[ridx * HEADS + head];
                const int ph2 = wh * WS + i2, pw2 = ww * WS + j2;
                const int reg2 = (ph2 < 49 ? 0 : (ph2 < 53 ? 1 : 2)) * 3
                               + (pw2 < 49 ? 0 : (pw2 < 53 ? 1 : 2));
                vals[ct] = sv[ct][i] + bv + (reg1 != reg2 ? -100.f : 0.f);
            } else {
                vals[ct] = -1e30f;
            }
        }
        float mx = fmaxf(fmaxf(vals[0], vals[1]), fmaxf(vals[2], vals[3]));
        #pragma unroll
        for (int d = 8; d >= 1; d >>= 1) mx = fmaxf(mx, __shfl_xor(mx, d, 32));
        float sm = 0.f;
        #pragma unroll
        for (int ct = 0; ct < 4; ++ct) { vals[ct] = __expf(vals[ct] - mx); sm += vals[ct]; }
        #pragma unroll
        for (int d = 8; d >= 1; d >>= 1) sm += __shfl_xor(sm, d, 32);
        const float inv = 1.f / sm;
        #pragma unroll
        for (int ct = 0; ct < 4; ++ct)
            Ps[wv][hi * 8 + i][16 * ct + r] = (_Float16)(vals[ct] * inv);
    }
    __syncthreads();

    v8f o0 = {}, o1 = {};
    #pragma unroll
    for (int ks = 0; ks < 2; ++ks) {
        v16h pa, vb0, vb1;
        #pragma unroll
        for (int e = 0; e < 16; ++e) {
            const int kk = kmap(e, hi);
            pa[e]  = Ps[wv][r][32 * ks + kk];
            vb0[e] = Vs[32 * ks + kk][r];
            vb1[e] = Vs[32 * ks + kk][16 + r];
        }
        o0 = __builtin_amdgcn_wmma_f32_16x16x32_f16(false, pa, false, vb0, (short)0, o0, false, false);
        o1 = __builtin_amdgcn_wmma_f32_16x16x32_f16(false, pa, false, vb1, (short)0, o1, false, false);
    }
    const int64_t ob = (int64_t)win * NPAD * DIM + head * HD;
    #pragma unroll
    for (int i = 0; i < 8; ++i) {
        const int n = rm + hi * 8 + i;
        aout[ob + (int64_t)n * DIM + r]      = (_Float16)o0[i];
        aout[ob + (int64_t)n * DIM + 16 + r] = (_Float16)o1[i];
    }
}

// ------------------------------- launcher ----------------------------------
extern "C" void kernel_launch(void* const* d_in, const int* in_sizes, int n_in,
                              void* d_out, int out_size, void* d_ws, size_t ws_size,
                              hipStream_t stream) {
    (void)in_sizes; (void)n_in; (void)out_size; (void)ws_size;
    const float* x      = (const float*)d_in[0];
    const float* g1     = (const float*)d_in[1];
    const float* b1     = (const float*)d_in[2];
    const float* qkv_w  = (const float*)d_in[3];
    const float* qkv_b  = (const float*)d_in[4];
    const float* rpb    = (const float*)d_in[5];
    const float* proj_w = (const float*)d_in[6];
    const float* proj_b = (const float*)d_in[7];
    const float* g2     = (const float*)d_in[8];
    const float* b2     = (const float*)d_in[9];
    const float* fc1_w  = (const float*)d_in[10];
    const float* fc1_b  = (const float*)d_in[11];
    const float* fc2_w  = (const float*)d_in[12];
    const float* fc2_b  = (const float*)d_in[13];
    float* out = (float*)d_out;
    char* ws = (char*)d_ws;

    auto align256 = [](size_t v) { return (v + 255) & ~(size_t)255; };
    size_t off = 0;
    const size_t oWqkv = off;  off = align256(off + (size_t)DIM * QKV_N * 2);
    const size_t oWproj = off; off = align256(off + (size_t)DIM * DIM * 2);
    const size_t oWfc1 = off;  off = align256(off + (size_t)DIM * MLP_H * 2);
    const size_t oWfc2 = off;  off = align256(off + (size_t)MLP_H * DIM * 2);
    // xwin -> (reused) attn-out -> (reused) h2
    const size_t oXWIN = off;  off = align256(off + (size_t)NWIN * NPAD * DIM * 2);
    // qkv -> (reused) mlp hidden  (sized for the larger of the two)
    const size_t qkvEl = (size_t)NWIN * NPAD * QKV_N;
    const size_t hidEl = (size_t)BL * MLP_H;
    const size_t oBIG = off;   off = align256(off + (qkvEl > hidEl ? qkvEl : hidEl) * 2);
    const size_t oX2 = off;    off = align256(off + (size_t)BL * DIM * 4);

    _Float16* wQKV  = (_Float16*)(ws + oWqkv);
    _Float16* wPROJ = (_Float16*)(ws + oWproj);
    _Float16* wFC1  = (_Float16*)(ws + oWfc1);
    _Float16* wFC2  = (_Float16*)(ws + oWfc2);
    _Float16* XWIN  = (_Float16*)(ws + oXWIN);
    _Float16* BIG   = (_Float16*)(ws + oBIG);
    float*    X2    = (float*)(ws + oX2);

    // K1: weight conversion
    auto cvtLaunch = [&](const float* s, _Float16* d, int n) {
        cvt_f32_f16<<<(n + 255) / 256, 256, 0, stream>>>(s, d, n);
    };
    cvtLaunch(qkv_w,  wQKV,  DIM * QKV_N);
    cvtLaunch(proj_w, wPROJ, DIM * DIM);
    cvtLaunch(fc1_w,  wFC1,  DIM * MLP_H);
    cvtLaunch(fc2_w,  wFC2,  MLP_H * DIM);

    // K2: LN1 + shift + window partition
    ln_kernel<<<(NWIN * NPAD) / 8, 256, 0, stream>>>(x, g1, b1, XWIN, 0, NWIN * NPAD);

    // K3: QKV GEMM  [131072 x 384] @ [384 x 1152]
    gemm_wmma<0><<<dim3(QKV_N / 64, (NWIN * NPAD) / 128), 256, 0, stream>>>(
        XWIN, DIM, wQKV, QKV_N, qkv_b, BIG, QKV_N, nullptr, nullptr, DIM);

    // K4: windowed attention (2048 windows x 12 heads)
    attn_kernel<<<NWIN * HEADS, 128, 0, stream>>>(BIG, rpb, XWIN);

    // K5: proj GEMM + window reverse + roll + residual -> X2 (f32)
    gemm_wmma<1><<<dim3(DIM / 64, (NWIN * NPAD) / 128), 256, 0, stream>>>(
        XWIN, DIM, wPROJ, DIM, proj_b, nullptr, 0, X2, x, DIM);

    // K6: LN2 -> h2 (reuse XWIN region)
    ln_kernel<<<BL / 8, 256, 0, stream>>>(X2, g2, b2, XWIN, 1, BL);

    // K7: fc1 + GELU -> hidden (reuse BIG region)
    gemm_wmma<2><<<dim3(MLP_H / 64, BL / 128), 256, 0, stream>>>(
        XWIN, DIM, wFC1, MLP_H, fc1_b, BIG, MLP_H, nullptr, nullptr, DIM);

    // K8: fc2 + residual -> d_out (f32)
    gemm_wmma<3><<<dim3(DIM / 64, BL / 128), 256, 0, stream>>>(
        BIG, MLP_H, wFC2, DIM, fc2_b, nullptr, 0, out, X2, MLP_H);
}
